// GatedScreeningTile_87265145520845
// MI455X (gfx1250) — matrix operand order repaired and checked by reading, not compiled, not run on hardware
//
#include <hip/hip_runtime.h>
#include <hip/hip_bf16.h>
#include <math.h>

// ---------------------------------------------------------------------------
// GatedScreeningTile for MI455X (gfx1250, wave32, WMMA + TDM)
//   B=1, N=4096, D=1024, H=8, DK=16, DV=64, window <= 257 (causal band)
// ---------------------------------------------------------------------------

typedef _Float16 f16;
typedef __attribute__((ext_vector_type(16))) _Float16 v16h;
typedef __attribute__((ext_vector_type(8)))  _Float16 v8h;
typedef __attribute__((ext_vector_type(8)))  float    v8f;
typedef __attribute__((ext_vector_type(4)))  unsigned int v4u;
typedef __attribute__((ext_vector_type(8)))  unsigned int v8u;

#define SEQ    4096
#define DMODEL 1024
#define NH     8
#define DKQ    16
#define DVH    64
#define CPROJ  640    // H*(DK+DV)
#define CINT   512    // H*DV

// LDS row stride (in halves) for the transposed B tile: 40 halves = 80 B.
// 80 B is 16 B aligned (clean ds_load_b128) and 20 dwords -> the 16 fragment
// columns land on 16 distinct banks (20*c mod 64 distinct for c=0..15).
#define LDSB_STRIDE 40

// ---------------------------------------------------------------------------
// WMMA GEMM: C(MxN fp32) = A(MxK fp32) * B(KxN fp32), all row-major.
// Block = 256 threads (8 waves), block tile 128(M) x 64(N), K step 32.
//   wave (mw = w>>1, nw = w&1) owns a 32x32 subtile = 2x2 WMMA tiles.
// A fragments: two contiguous 8-float runs per lane -> float4 loads + cvt_pk.
// B tile: cooperatively staged to LDS *transposed* f16; fragment = two 8-half
// runs per lane (k 8h+e, 16+8h+e) -> ds_load_b128 pairs.
// Requires: M % 128 == 0, N % 64 == 0, K % 32 == 0 (true for all uses).
// ---------------------------------------------------------------------------
__global__ __launch_bounds__(256) void gemm_wmma_f16acc32(
    const float* __restrict__ A, const float* __restrict__ B,
    float* __restrict__ C, int M, int N, int K)
{
    __shared__ _Float16 ldsBt[64 * LDSB_STRIDE];   // 5 KB

    const int lane = threadIdx.x & 31;
    const int wave = threadIdx.x >> 5;
    const int col  = lane & 15;
    const int half = lane >> 4;

    const int bnCount = N >> 6;
    const int bm = blockIdx.x / bnCount;
    const int bn = blockIdx.x % bnCount;
    const int n0blk = bn << 6;                      // block N origin
    const int m0w = (bm << 7) + ((wave >> 1) << 5); // wave M origin (32 rows)
    const int n0w = ((wave & 1) << 5);              // wave N origin in block

    v8f acc[2][2] = {{{}, {}}, {{}, {}}};

    for (int k = 0; k < K; k += 32) {
        // ---- stage B tile (32 x 64 f32) transposed into LDS as f16 --------
#pragma unroll
        for (int r = 0; r < 8; ++r) {
            const int idx = r * 256 + threadIdx.x;  // 2048 elements
            const int nn = idx & 63;
            const int kk = idx >> 6;
            ldsBt[nn * LDSB_STRIDE + kk] = (f16)B[(k + kk) * N + n0blk + nn];
        }
        __syncthreads();

        // ---- A fragments: direct vectorized global loads ------------------
        v16h af[2];
#pragma unroll
        for (int mt = 0; mt < 2; ++mt) {
            const int rowIdx = m0w + mt * 16 + col;
            const float4* a4 = (const float4*)(A + (size_t)rowIdx * K + k);
            __builtin_prefetch(A + (size_t)rowIdx * K + k + 32, 0, 0);
            const float4 f0 = a4[2 * half];
            const float4 f1 = a4[2 * half + 1];
            const float4 f2 = a4[4 + 2 * half];
            const float4 f3 = a4[5 + 2 * half];
            af[mt][0]  = (f16)f0.x; af[mt][1]  = (f16)f0.y;
            af[mt][2]  = (f16)f0.z; af[mt][3]  = (f16)f0.w;
            af[mt][4]  = (f16)f1.x; af[mt][5]  = (f16)f1.y;
            af[mt][6]  = (f16)f1.z; af[mt][7]  = (f16)f1.w;
            af[mt][8]  = (f16)f2.x; af[mt][9]  = (f16)f2.y;
            af[mt][10] = (f16)f2.z; af[mt][11] = (f16)f2.w;
            af[mt][12] = (f16)f3.x; af[mt][13] = (f16)f3.y;
            af[mt][14] = (f16)f3.z; af[mt][15] = (f16)f3.w;
        }

        // ---- B fragments from transposed LDS (wide ds reads) --------------
        v16h bf[2];
#pragma unroll
        for (int nt = 0; nt < 2; ++nt) {
            const int cbase = (n0w + nt * 16 + col) * LDSB_STRIDE + 8 * half;
            const v8h lo = *(const v8h*)&ldsBt[cbase];
            const v8h hi = *(const v8h*)&ldsBt[cbase + 16];
#pragma unroll
            for (int e = 0; e < 8; ++e) { bf[nt][e] = lo[e]; bf[nt][8 + e] = hi[e]; }
        }

#pragma unroll
        for (int mt = 0; mt < 2; ++mt)
#pragma unroll
            for (int nt = 0; nt < 2; ++nt)
                acc[mt][nt] = __builtin_amdgcn_wmma_f32_16x16x32_f16(
                    false, af[mt], false, bf[nt], (short)0, acc[mt][nt],
                    false, false);
        __syncthreads();
    }

    // ---- epilogue: C layout (vgpr r -> M = r + 8*half, N = col) ----------
#pragma unroll
    for (int mt = 0; mt < 2; ++mt)
#pragma unroll
        for (int nt = 0; nt < 2; ++nt)
#pragma unroll
            for (int r = 0; r < 8; ++r)
                C[(size_t)(m0w + mt * 16 + r + (half << 3)) * N
                  + n0blk + n0w + nt * 16 + col] = acc[mt][nt][r];
}

// ---------------------------------------------------------------------------
// Split qg/kv, L2-normalize q,k,v, precompute gate = tanh(silu(g)).
// ---------------------------------------------------------------------------
__global__ void norm_split_kernel(
    const float* __restrict__ qg, const float* __restrict__ kv,
    f16* __restrict__ qn, f16* __restrict__ kn, f16* __restrict__ vn,
    float* __restrict__ gate)
{
    const int t = blockIdx.x * blockDim.x + threadIdx.x;
    if (t >= SEQ * NH) return;
    const int n = t >> 3, h = t & 7;
    const float* qrow = qg + n * CPROJ + h * 80;       // q[0:16], g[16:80]
    const float* krow = kv + n * CPROJ + h * 80;       // k[0:16], v[16:80]

    float ss = 0.f;
    for (int i = 0; i < DKQ; ++i) ss += qrow[i] * qrow[i];
    float inv = 1.f / fmaxf(sqrtf(ss), 1e-12f);
    for (int i = 0; i < DKQ; ++i)
        qn[n * (NH * DKQ) + h * DKQ + i] = (f16)(qrow[i] * inv);

    ss = 0.f;
    for (int i = 0; i < DKQ; ++i) ss += krow[i] * krow[i];
    inv = 1.f / fmaxf(sqrtf(ss), 1e-12f);
    for (int i = 0; i < DKQ; ++i)
        kn[n * (NH * DKQ) + h * DKQ + i] = (f16)(krow[i] * inv);

    const float* vrow = krow + DKQ;
    ss = 0.f;
    for (int i = 0; i < DVH; ++i) ss += vrow[i] * vrow[i];
    inv = 1.f / fmaxf(sqrtf(ss), 1e-12f);
    for (int i = 0; i < DVH; ++i)
        vn[n * CINT + h * DVH + i] = (f16)(vrow[i] * inv);

    const float* grow = qrow + DKQ;
    for (int i = 0; i < DVH; ++i) {
        const float x = grow[i];
        const float silu = x / (1.f + expf(-x));
        gate[n * CINT + h * DVH + i] = tanhf(silu);
    }
}

// ---------------------------------------------------------------------------
// Banded attention. One wave per (head, 16-row query block); no block-level
// barriers (per-wave LDS regions only -> safe with wave-uniform branches).
// Band: j in [i0-272, i0+15] -> 9 key-block pairs of 32.
// v tiles (32x64 f16, row stride 512) are pulled into LDS by the Tensor Data
// Mover (tensor_load_to_lds, 2D D#), completion via s_wait_tensorcnt; the
// j0<0 edge blocks fall back to masked global loads (wave-uniform branch).
// ---------------------------------------------------------------------------
__global__ __launch_bounds__(128) void attn_band_wmma(
    const f16* __restrict__ qn, const f16* __restrict__ kn,
    const f16* __restrict__ vn, const float* __restrict__ gate,
    const float* __restrict__ log_window, const float* __restrict__ log_r,
    const float* __restrict__ log_hscale, float* __restrict__ inter)
{
    __shared__ _Float16 attnLds[4][16 * 32];   // per-wave 16x32 attn tile
    __shared__ _Float16 vLds[4][32 * 64];      // per-wave TDM v tile (4 KB)

    const int lane = threadIdx.x & 31;
    const int wave = threadIdx.x >> 5;
    const int h  = blockIdx.y;
    const int qb = blockIdx.x * 4 + wave;
    const int i0 = qb << 4;
    const int col  = lane & 15;
    const int half = lane >> 4;

    const float w   = expf(log_window[h]) + 1.f;
    const float rr_ = expf(log_r[h]) + 1.f;
    const float hs  = expf(log_hscale[h]);

    _Float16* vLdsW = &vLds[wave][0];
    const unsigned ldsVOff = (unsigned)(unsigned long long)(void*)vLdsW;

    // q fragment (A-layout), DK=16 padded to K=32: per-lane contiguous v8h.
    v16h aq = {};
    {
        const v8h q0 = *(const v8h*)&qn[(size_t)(i0 + col) * (NH * DKQ)
                                        + h * DKQ + 8 * half];
#pragma unroll
        for (int e = 0; e < 8; ++e) aq[e] = q0[e];   // kk >= 16 stays zero
    }

    v8f acc[4] = {{}, {}, {}, {}};

    for (int p = 0; p < 9; ++p) {
        const int j0 = i0 - 272 + 32 * p;    // pair covers keys j0 .. j0+31
        const bool tdm = (j0 >= 0);          // wave-uniform

        if (tdm) {
            // WAR: TDM engine is not ordered with this wave's DS pipe --
            // drain outstanding LDS reads of the previous tile first.
            asm volatile("s_wait_dscnt 0" ::: "memory");

            const unsigned long long ga =
                (unsigned long long)(const void*)(vn + (size_t)j0 * CINT
                                                  + h * DVH);
            // D# group0: count=1 | lds_addr | global_addr(57b) | type=2
            v4u g0;
            g0[0] = 1u;
            g0[1] = ldsVOff;
            g0[2] = (unsigned)ga;
            g0[3] = (unsigned)((ga >> 32) & 0x01FFFFFFu) | (2u << 30);
            // D# group1: mask=0, data_size=1(2B), dims/tile/stride (2D)
            v8u g1;
            g1[0] = (1u << 16);          // data_size = 2 bytes
            g1[1] = (64u << 16);         // tensor_dim0[15:0] = 64
            g1[2] = (32u << 16);         // dim0[31:16]=0 | tensor_dim1[15:0]=32
            g1[3] = (64u << 16);         // dim1[31:16]=0 | tile_dim0 = 64
            g1[4] = 32u;                 // tile_dim1 = 32 | tile_dim2 = 0
            g1[5] = (unsigned)CINT;      // tensor_dim0_stride = 512 elements
            g1[6] = 0u;
            g1[7] = 0u;
            asm volatile("tensor_load_to_lds %0, %1"
                         :: "s"(g0), "s"(g1) : "memory");
            __builtin_amdgcn_s_wait_tensorcnt(0);
        }

        // --- similarity + windowed gating for both 16-key sub-blocks -------
#pragma unroll
        for (int sb = 0; sb < 2; ++sb) {
            const int js = j0 + 16 * sb;
            const int j  = js + col;
            // B fragment: lane col = key index; elements 0..7 carry
            // K = 8*half + e (DK=16), elements 8..15 are K-padding zeros.
            v16h bk = {};
            if (j >= 0) {
                const f16* kr = &kn[(size_t)j * (NH * DKQ) + h * DKQ];
                const v8h k0 = *(const v8h*)(kr + 8 * half);
#pragma unroll
                for (int e = 0; e < 8; ++e) { bk[e] = k0[e]; }
            }
            v8f zero = {};
            v8f sim = __builtin_amdgcn_wmma_f32_16x16x32_f16(
                false, aq, false, bk, (short)0, zero, false, false);

#pragma unroll
            for (int r = 0; r < 8; ++r) {
                const int i = i0 + r + (half << 3);
                const float dist = (float)(j - i);
                float a = 0.f;
                if (j >= 0 && dist <= 0.f && dist > -w) {
                    float tt = 1.f - rr_ * (1.f - sim[r]);
                    tt = fmaxf(tt, 0.f);
                    const float soft =
                        0.5f * (cosf(dist * 3.14159265358979323846f / w) + 1.f);
                    a = tt * tt * soft;
                }
                attnLds[wave][(r + (half << 3)) * 32 + 16 * sb + col] = (f16)a;
            }
        }

        // --- aggr += attn(16x32) x v(32x64), 4 N-tiles ---------------------
        // attn tile re-read in A-layout (per-wave DS ops are in-order).
        v16h aa;
#pragma unroll
        for (int e = 0; e < 16; ++e) {
            const int v = e >> 1, s = e & 1;
            const int kk = (half << 4) + e;
            aa[e] = attnLds[wave][col * 32 + kk];
        }
#pragma unroll
        for (int nt = 0; nt < 4; ++nt) {
            v16h bv;
            if (tdm) {
#pragma unroll
                for (int e = 0; e < 16; ++e) {
                    const int kk = (half << 4) + e;
                    bv[e] = vLdsW[kk * 64 + nt * 16 + col];
                }
            } else {
#pragma unroll
                for (int e = 0; e < 16; ++e) {
                    const int kk = (half << 4) + e;
                    const int j  = j0 + kk;
                    const int jc = j < 0 ? 0 : j;
                    bv[e] = (j >= 0)
                        ? vn[(size_t)jc * CINT + h * DVH + nt * 16 + col]
                        : (f16)0;
                }
            }
            acc[nt] = __builtin_amdgcn_wmma_f32_16x16x32_f16(
                false, aa, false, bv, (short)0, acc[nt], false, false);
        }
    }

    // --- epilogue: per-row L2 norm over DV=64, tanh normalize, gate, scale -
#pragma unroll
    for (int r = 0; r < 8; ++r) {
        float ss = 0.f;
#pragma unroll
        for (int nt = 0; nt < 4; ++nt) ss += acc[nt][r] * acc[nt][r];
        ss += __shfl_xor(ss, 1, 32);
        ss += __shfl_xor(ss, 2, 32);
        ss += __shfl_xor(ss, 4, 32);
        ss += __shfl_xor(ss, 8, 32);
        const float nrm   = sqrtf(ss);
        const float scale = tanhf(nrm) / fmaxf(nrm, 1e-12f);
        const int i = i0 + r + (half << 3);
#pragma unroll
        for (int nt = 0; nt < 4; ++nt) {
            const int d = nt * 16 + col;
            inter[(size_t)i * CINT + h * DVH + d] =
                acc[nt][r] * scale * gate[(size_t)i * CINT + h * DVH + d] * hs;
        }
    }
}

// ---------------------------------------------------------------------------
extern "C" void kernel_launch(void* const* d_in, const int* in_sizes, int n_in,
                              void* d_out, int out_size, void* d_ws, size_t ws_size,
                              hipStream_t stream)
{
    const float* tokens     = (const float*)d_in[0];   // (1,4096,1024)
    const float* w_qg       = (const float*)d_in[1];   // (1024,8,80)
    const float* w_kv       = (const float*)d_in[2];   // (1024,8,80)
    const float* w_out      = (const float*)d_in[3];   // (512,1024)
    const float* log_window = (const float*)d_in[4];   // (8,)
    const float* log_r      = (const float*)d_in[5];   // (8,)
    const float* log_hscale = (const float*)d_in[6];   // (8,)
    float* out = (float*)d_out;                        // (1,4096,1024) fp32

    // workspace carve-up (~44 MB)
    char* ws = (char*)d_ws;
    float* qg   = (float*)ws;  ws += (size_t)SEQ * CPROJ * sizeof(float);
    float* kv   = (float*)ws;  ws += (size_t)SEQ * CPROJ * sizeof(float);
    f16*  qn    = (f16*)ws;    ws += (size_t)SEQ * NH * DKQ * sizeof(f16);
    f16*  kn    = (f16*)ws;    ws += (size_t)SEQ * NH * DKQ * sizeof(f16);
    f16*  vn    = (f16*)ws;    ws += (size_t)SEQ * CINT * sizeof(f16);
    float* gate = (float*)ws;  ws += (size_t)SEQ * CINT * sizeof(float);
    float* inter= (float*)ws;  ws += (size_t)SEQ * CINT * sizeof(float);

    // 1+2) input projections: 4096x1024 * 1024x640 (block tile 128x64)
    gemm_wmma_f16acc32<<<(SEQ / 128) * (CPROJ / 64), 256, 0, stream>>>(
        tokens, w_qg, qg, SEQ, CPROJ, DMODEL);
    gemm_wmma_f16acc32<<<(SEQ / 128) * (CPROJ / 64), 256, 0, stream>>>(
        tokens, w_kv, kv, SEQ, CPROJ, DMODEL);

    // 3) split + l2norm + gate precompute
    norm_split_kernel<<<(SEQ * NH + 255) / 256, 256, 0, stream>>>(
        qg, kv, qn, kn, vn, gate);

    // 4) banded attention (WMMA + TDM), wave per (head, 16-query block)
    {
        dim3 grid(SEQ / 16 / 4, NH);       // (64, 8), 4 waves/block
        attn_band_wmma<<<grid, 128, 0, stream>>>(
            qn, kn, vn, gate, log_window, log_r, log_hscale, inter);
    }

    // 5) output projection: 4096x512 * 512x1024 -> d_out
    gemm_wmma_f16acc32<<<(SEQ / 128) * (DMODEL / 64), 256, 0, stream>>>(
        inter, w_out, out, SEQ, DMODEL, CINT);
}
